// MultiQueryAttention_59983513256625
// MI455X (gfx1250) — matrix operand order repaired and checked by reading, not compiled, not run on hardware
//
#include <hip/hip_runtime.h>
#include <hip/hip_bf16.h>
#include <cstdint>

// ---------------------------------------------------------------------------
// MQA forward for B=2, S=2048, D=2048, H=16, d=128 on gfx1250 (wave32, WMMA)
// ---------------------------------------------------------------------------

typedef __attribute__((ext_vector_type(16))) __bf16 v16bf;
typedef __attribute__((ext_vector_type(8)))  float  v8f;

union Frag {
    v16bf v;
    uint4 q[2];
};

static __device__ __forceinline__ uint16_t f32_to_bf16_bits(float f) {
    uint32_t u = __float_as_uint(f);
    u = (u + 0x7FFFu + ((u >> 16) & 1u)) >> 16;   // round-to-nearest-even
    return (uint16_t)u;
}

static __device__ __forceinline__ v8f wmma_bf16(const Frag& a, const Frag& b, v8f c) {
    return __builtin_amdgcn_wmma_f32_16x16x32_bf16(
        /*neg_a=*/false, a.v, /*neg_b=*/false, b.v,
        /*c_mod=*/(short)0, c, /*reuse_a=*/false, /*reuse_b=*/false);
}

// -------------------------- elementwise converters --------------------------

__global__ void cvt_f32_bf16(const float* __restrict__ src,
                             uint16_t* __restrict__ dst, int n) {
    int i = blockIdx.x * blockDim.x + threadIdx.x;
    if (i < n) dst[i] = f32_to_bf16_bits(src[i]);
}

// Wt[n*K + k] = bf16(W[k*N + n])   (row-major [K,N] -> row-major [N,K])
__global__ void cvt_transpose_f32_bf16(const float* __restrict__ W,
                                       uint16_t* __restrict__ Wt, int K, int N) {
    int i = blockIdx.x * blockDim.x + threadIdx.x;
    if (i < K * N) {
        int n = i / K, k = i - n * K;
        Wt[i] = f32_to_bf16_bits(W[(size_t)k * N + n]);
    }
}

// vT[b*128*2048 + d*2048 + s] = vb[(b*2048+s)*128 + d]
__global__ void transpose_v_bf16(const uint16_t* __restrict__ vb,
                                 uint16_t* __restrict__ vT) {
    int i = blockIdx.x * blockDim.x + threadIdx.x;   // 524288 total
    int b = i >> 18;
    int r = i & 0x3FFFF;
    int d = r >> 11;
    int s = r & 2047;
    vT[i] = vb[((size_t)(b * 2048 + s)) * 128 + d];
}

// --------------------- WMMA GEMM, 32x32 wave tile --------------------------
// C[M,N] = A[M,K] * Bt[N,K]^T + bias[N].  Block = 64x64 (4 waves, 32x32 each).
// Used for the skinny K/V projections (N=128) to keep grid parallelism.
__global__ void gemm_bf16_wmma32(const uint16_t* __restrict__ A,
                                 const uint16_t* __restrict__ Bt,
                                 const float* __restrict__ bias,
                                 uint16_t* __restrict__ Cb,
                                 int M, int N, int K) {
    const int lane = threadIdx.x & 31;
    const int wave = threadIdx.x >> 5;
    const int half = lane >> 4;
    const int lr   = lane & 15;
    const int m0 = blockIdx.x * 64 + (wave & 1) * 32;
    const int n0 = blockIdx.y * 64 + (wave >> 1) * 32;

    v8f acc[2][2] = {};

    for (int k0 = 0; k0 < K; k0 += 32) {
        Frag a[2], b[2];
#pragma unroll
        for (int t = 0; t < 2; ++t) {
            const uint16_t* ap = A + (size_t)(m0 + t * 16 + lr) * K + k0 + half * 8;
            a[t].q[0] = *(const uint4*)ap;
            a[t].q[1] = *(const uint4*)(ap + 16);
            const uint16_t* bp = Bt + (size_t)(n0 + t * 16 + lr) * K + k0 + half * 8;
            b[t].q[0] = *(const uint4*)bp;
            b[t].q[1] = *(const uint4*)(bp + 16);
        }
#pragma unroll
        for (int mt = 0; mt < 2; ++mt)
#pragma unroll
            for (int nt = 0; nt < 2; ++nt)
                acc[mt][nt] = wmma_bf16(a[mt], b[nt], acc[mt][nt]);
    }

#pragma unroll
    for (int mt = 0; mt < 2; ++mt)
#pragma unroll
        for (int nt = 0; nt < 2; ++nt) {
            const int col = n0 + nt * 16 + lr;
            const float bv = bias[col];
#pragma unroll
            for (int r = 0; r < 8; ++r) {
                const int row = m0 + mt * 16 + r + half * 8;
                Cb[(size_t)row * N + col] = f32_to_bf16_bits(acc[mt][nt][r] + bv);
            }
        }
}

// --------------------- WMMA GEMM, 64x64 wave tile --------------------------
// Block = 128x128 (4 waves in 2x2, 64x64 each).  16 WMMA per 32-K step,
// 8 KB loaded per step -> ~32 FLOP/byte from L2 (2x the 32x32 tile).
// Cb (bf16) and/or Cf (f32) output.
__global__ void gemm_bf16_wmma64(const uint16_t* __restrict__ A,
                                 const uint16_t* __restrict__ Bt,
                                 const float* __restrict__ bias,
                                 uint16_t* __restrict__ Cb,
                                 float* __restrict__ Cf,
                                 int M, int N, int K) {
    const int lane = threadIdx.x & 31;
    const int wave = threadIdx.x >> 5;
    const int half = lane >> 4;
    const int lr   = lane & 15;
    const int m0 = blockIdx.x * 128 + (wave & 1) * 64;
    const int n0 = blockIdx.y * 128 + (wave >> 1) * 64;

    v8f acc[4][4] = {};

    for (int k0 = 0; k0 < K; k0 += 32) {
        Frag a[4], b[4];
#pragma unroll
        for (int t = 0; t < 4; ++t) {
            const uint16_t* ap = A + (size_t)(m0 + t * 16 + lr) * K + k0 + half * 8;
            a[t].q[0] = *(const uint4*)ap;
            a[t].q[1] = *(const uint4*)(ap + 16);
            const uint16_t* bp = Bt + (size_t)(n0 + t * 16 + lr) * K + k0 + half * 8;
            b[t].q[0] = *(const uint4*)bp;
            b[t].q[1] = *(const uint4*)(bp + 16);
        }
#pragma unroll
        for (int mt = 0; mt < 4; ++mt)
#pragma unroll
            for (int nt = 0; nt < 4; ++nt)
                acc[mt][nt] = wmma_bf16(a[mt], b[nt], acc[mt][nt]);
    }

#pragma unroll
    for (int mt = 0; mt < 4; ++mt)
#pragma unroll
        for (int nt = 0; nt < 4; ++nt) {
            const int col = n0 + nt * 16 + lr;
            const float bv = bias[col];
#pragma unroll
            for (int r = 0; r < 8; ++r) {
                const int row = m0 + mt * 16 + r + half * 8;
                const float val = acc[mt][nt][r] + bv;
                if (Cb) Cb[(size_t)row * N + col] = f32_to_bf16_bits(val);
                if (Cf) Cf[(size_t)row * N + col] = val;
            }
        }
}

// --------------------------- Flash MQA attention ---------------------------
// One wave handles 16 query rows of one (b, h); streams K/V in 32-col blocks.
// qb: [B*S, 2048] bf16;  kb: [B*S, 128] bf16;  vT: [B][128][2048] bf16
// attn_out: [B*S, 2048] bf16
__global__ void mqa_flash_attn(const uint16_t* __restrict__ qb,
                               const uint16_t* __restrict__ kb,
                               const uint16_t* __restrict__ vT,
                               uint16_t* __restrict__ attn_out) {
    __shared__ __align__(16) uint16_t Pb[4][16 * 32];   // per-wave P staging

    const int lane = threadIdx.x & 31;
    const int wave = threadIdx.x >> 5;
    const int half = lane >> 4;
    const int lr   = lane & 15;

    const int gw    = blockIdx.x * 4 + wave;       // 0..4095
    const int qtile = gw & 127;                    // S/16
    const int h     = (gw >> 7) & 15;
    const int b     = gw >> 11;
    const int q0    = qtile * 16;

    const float scale = 0.08838834764831845f;      // 1/sqrt(128)

    // Preload Q fragments: 16 rows x 128 d, as 4 A-frags of 16x32
    Frag qa[4];
    {
        const uint16_t* qrow = qb + ((size_t)(b * 2048 + q0 + lr)) * 2048 + h * 128;
#pragma unroll
        for (int ds = 0; ds < 4; ++ds) {
            const uint16_t* p = qrow + ds * 32 + half * 8;
            qa[ds].q[0] = *(const uint4*)p;
            qa[ds].q[1] = *(const uint4*)(p + 16);
        }
    }

    v8f acco[8] = {};
    float mrow[8], lrow[8];
#pragma unroll
    for (int r = 0; r < 8; ++r) { mrow[r] = -1e30f; lrow[r] = 0.0f; }

    const uint16_t* kbase = kb + (size_t)b * 2048 * 128;
    const uint16_t* vbase = vT + (size_t)b * 128 * 2048;
    uint16_t* Pw = Pb[wave];

    for (int j0 = 0; j0 < 2048; j0 += 32) {
        // ---- scores: two 16x16 tiles over this 32-column K block ----
        v8f s0 = {}, s1 = {};
#pragma unroll
        for (int ds = 0; ds < 4; ++ds) {
            Frag bk0, bk1;
            const uint16_t* kp0 = kbase + (size_t)(j0 + lr) * 128 + ds * 32 + half * 8;
            bk0.q[0] = *(const uint4*)kp0;
            bk0.q[1] = *(const uint4*)(kp0 + 16);
            const uint16_t* kp1 = kbase + (size_t)(j0 + 16 + lr) * 128 + ds * 32 + half * 8;
            bk1.q[0] = *(const uint4*)kp1;
            bk1.q[1] = *(const uint4*)(kp1 + 16);
            s0 = wmma_bf16(qa[ds], bk0, s0);
            s1 = wmma_bf16(qa[ds], bk1, s1);
        }

        // ---- online softmax update (row stats across 16-lane halves) ----
#pragma unroll
        for (int r = 0; r < 8; ++r) {
            float a0 = s0[r] * scale;
            float a1 = s1[r] * scale;
            float mx = fmaxf(a0, a1);
#pragma unroll
            for (int off = 1; off < 16; off <<= 1)
                mx = fmaxf(mx, __shfl_xor(mx, off, 32));
            const float mn = fmaxf(mrow[r], mx);
            const float alpha = __expf(mrow[r] - mn);
            const float p0 = __expf(a0 - mn);
            const float p1 = __expf(a1 - mn);
            float sum = p0 + p1;
#pragma unroll
            for (int off = 1; off < 16; off <<= 1)
                sum += __shfl_xor(sum, off, 32);
            lrow[r] = lrow[r] * alpha + sum;
            mrow[r] = mn;
#pragma unroll
            for (int dt = 0; dt < 8; ++dt) acco[dt][r] *= alpha;
            // stage P (C layout -> row-major LDS) for A-frag reload
            Pw[(r + half * 8) * 32 + lr]      = f32_to_bf16_bits(p0);
            Pw[(r + half * 8) * 32 + 16 + lr] = f32_to_bf16_bits(p1);
        }

        asm volatile("s_wait_dscnt 0" ::: "memory");

        // reload P as 16x32 A-fragment
        Frag pf;
        {
            const uint16_t* pp = Pw + lr * 32 + half * 8;
            pf.q[0] = *(const uint4*)pp;
            pf.q[1] = *(const uint4*)(pp + 16);
        }

        // ---- acc_o += P @ V_block  (8 d-tiles of 16 cols) ----
#pragma unroll
        for (int dt = 0; dt < 8; ++dt) {
            Frag vf;
            const uint16_t* vp = vbase + (size_t)(dt * 16 + lr) * 2048 + j0 + half * 8;
            vf.q[0] = *(const uint4*)vp;
            vf.q[1] = *(const uint4*)(vp + 16);
            acco[dt] = wmma_bf16(pf, vf, acco[dt]);
        }
    }

    // ---- epilogue: normalize, store bf16 ----
#pragma unroll
    for (int dt = 0; dt < 8; ++dt)
#pragma unroll
        for (int r = 0; r < 8; ++r) {
            const int row = q0 + r + half * 8;
            const float val = acco[dt][r] / lrow[r];
            attn_out[((size_t)(b * 2048 + row)) * 2048 + h * 128 + dt * 16 + lr] =
                f32_to_bf16_bits(val);
        }
}

// ------------------------------- launcher ----------------------------------

extern "C" void kernel_launch(void* const* d_in, const int* in_sizes, int n_in,
                              void* d_out, int out_size, void* d_ws, size_t ws_size,
                              hipStream_t stream) {
    const float* x  = (const float*)d_in[0];
    const float* Wq = (const float*)d_in[1];
    const float* bq = (const float*)d_in[2];
    const float* Wk = (const float*)d_in[3];
    const float* bk = (const float*)d_in[4];
    const float* Wv = (const float*)d_in[5];
    const float* bv = (const float*)d_in[6];
    const float* Wo = (const float*)d_in[7];
    const float* bo = (const float*)d_in[8];
    float* out = (float*)d_out;

    // workspace carve-up (bf16 elements)
    uint16_t* xb  = (uint16_t*)d_ws;        // 2*2048*2048 = 8388608
    uint16_t* Wqt = xb  + 8388608;          // 2048*2048  = 4194304
    uint16_t* Wkt = Wqt + 4194304;          // 128*2048   = 262144
    uint16_t* Wvt = Wkt + 262144;           // 262144
    uint16_t* Wot = Wvt + 262144;           // 4194304
    uint16_t* qb  = Wot + 4194304;          // 8388608
    uint16_t* kb  = qb  + 8388608;          // 2*2048*128 = 524288
    uint16_t* vb  = kb  + 524288;           // 524288
    uint16_t* vT  = vb  + 524288;           // 524288
    uint16_t* ao  = vT  + 524288;           // 8388608

    // 1) precision conversion (+ weight transpose for contiguous B-frags)
    cvt_f32_bf16<<<8388608 / 256, 256, 0, stream>>>(x, xb, 8388608);
    cvt_transpose_f32_bf16<<<4194304 / 256, 256, 0, stream>>>(Wq, Wqt, 2048, 2048);
    cvt_transpose_f32_bf16<<<262144 / 256, 256, 0, stream>>>(Wk, Wkt, 2048, 128);
    cvt_transpose_f32_bf16<<<262144 / 256, 256, 0, stream>>>(Wv, Wvt, 2048, 128);
    cvt_transpose_f32_bf16<<<4194304 / 256, 256, 0, stream>>>(Wo, Wot, 2048, 2048);

    // 2) projections: q = x Wq + bq ; k = x Wk + bk ; v = x Wv + bv
    dim3 blk(128);
    gemm_bf16_wmma64<<<dim3(32, 16), blk, 0, stream>>>(xb, Wqt, bq, qb, nullptr, 4096, 2048, 2048);
    gemm_bf16_wmma32<<<dim3(64, 2),  blk, 0, stream>>>(xb, Wkt, bk, kb, 4096, 128, 2048);
    gemm_bf16_wmma32<<<dim3(64, 2),  blk, 0, stream>>>(xb, Wvt, bv, vb, 4096, 128, 2048);

    // 3) V transpose for contiguous V B-frags in the AV matmul
    transpose_v_bf16<<<524288 / 256, 256, 0, stream>>>(vb, vT);

    // 4) flash MQA attention: 4096 waves (2 batch x 16 heads x 128 q-tiles)
    mqa_flash_attn<<<1024, 128, 0, stream>>>(qb, kb, vT, ao);

    // 5) output projection to f32: out = ao Wo + bo
    gemm_bf16_wmma64<<<dim3(32, 16), blk, 0, stream>>>(ao, Wot, bo, nullptr, out, 4096, 2048, 2048);
}
